// LargeJointGroupDecoder_53274774339864
// MI455X (gfx1250) — compile-verified, matrix operand order checked by reading, not run on hardware
//
#include <hip/hip_runtime.h>
#include <stdint.h>

// ---------------------------------------------------------------------------
// LargeJointGroupDecoder fused kernel for gfx1250 (MI455X).
//   B=32768, H=512, L=5.  Memory-bound: 256MB activation read dominates
//   (~11us at 23.3 TB/s).  Weights pre-converted to bf16 [N,K] in d_ws
//   (~20MB, L2-resident).  All big GEMMs via v_wmma_f32_16x16x32_bf16.
//   Explicit LDS/global address spaces + hoisted lane bases so fragment
//   loads are ds_load_b128 / global_load_b128 with immediate offsets.
// ---------------------------------------------------------------------------

#define MT    32      // batch rows per block
#define HDIM  512
#define TPB   256     // 8 waves of 32

#define AS3 __attribute__((address_space(3)))
#define AS1 __attribute__((address_space(1)))

typedef __attribute__((ext_vector_type(16))) __bf16    v16bf;
typedef __attribute__((ext_vector_type(8)))  float     v8f;
typedef __attribute__((ext_vector_type(4)))  float     v4f;
typedef __attribute__((ext_vector_type(4)))  uint32_t  v4u;

typedef AS3 uint16_t*        lds_u16;
typedef const AS3 uint16_t*  lds_cu16;
typedef const AS3 v4u*       lds_cv4u;
typedef const AS1 uint16_t*  glb_cu16;
typedef const AS1 float*     glb_cf32;
typedef const AS1 v4f*       glb_cv4f;
typedef AS1 float*           glb_f32;

union FragCvt { v4u u[2]; v16bf bf; };

__device__ __forceinline__ uint16_t f2bf(float x) {
  uint32_t u = __float_as_uint(x);
  uint32_t r = u + 0x7FFFu + ((u >> 16) & 1u);   // round-to-nearest-even
  return (uint16_t)(r >> 16);
}
__device__ __forceinline__ float bf2f(uint16_t h) {
  return __uint_as_float(((uint32_t)h) << 16);
}

// 32B fragment load from LDS (ds_load_b128 x2, immediate offsets off the base)
__device__ __forceinline__ v16bf ld_lds32(lds_cu16 p) {
  FragCvt fc;
  fc.u[0] = ((lds_cv4u)p)[0];
  fc.u[1] = ((lds_cv4u)p)[1];
  return fc.bf;
}
// 32B fragment load from global (global_load_b128 x2)
__device__ __forceinline__ v16bf ld_glb32(glb_cu16 p) {
  FragCvt fc;
  fc.u[0] = ((const AS1 v4u*)p)[0];
  fc.u[1] = ((const AS1 v4u*)p)[1];
  return fc.bf;
}

// One K=512 panel: acc[2][4] += A[MT x 512] @ Wt[:, kWbase..kWbase+512).
// A fragment (16Mx32K): lanes 0-15 rows m0.., K k0..k0+15; lanes 16-31 K k0+16..
// B fragment (32Kx16N): lane = output column, 16 contiguous K per lane.
// All per-lane bases hoisted; k-steps use immediate offsets.
__device__ __forceinline__ void gemm_panel(v8f acc[2][4], lds_cu16 A,
                                           glb_cu16 Wt, int Ktot,
                                           int kWbase, int n0, int lane) {
  const int lm = lane & 15;
  const int ls = (lane >> 4) << 4;                 // 0 or 16 (K half-select)
  lds_cu16 a0p = A + (lm * HDIM + ls);             // rows 0..15
  lds_cu16 a1p = a0p + 16 * HDIM;                  // rows 16..31
  glb_cu16 bp  = Wt + (size_t)((n0 + lm) * Ktot + kWbase + ls);
#pragma unroll 4
  for (int k0 = 0; k0 < HDIM; k0 += 32) {
    v16bf a0 = ld_lds32(a0p + k0);
    v16bf a1 = ld_lds32(a1p + k0);
#pragma unroll
    for (int j = 0; j < 4; ++j) {
      v16bf b = ld_glb32(bp + (size_t)(16 * j) * Ktot + k0);
      acc[0][j] = __builtin_amdgcn_wmma_f32_16x16x32_bf16(
          false, a0, false, b, (short)0, acc[0][j], false, false);
      acc[1][j] = __builtin_amdgcn_wmma_f32_16x16x32_bf16(
          false, a1, false, b, (short)0, acc[1][j], false, false);
    }
  }
}

// Epilogue over the 16x16 f32 C layout: VGPR r -> row r (lanes 0-15) / r+8
// (lanes 16-31); N = lane&15 within tile.
//  MODE 0: dst += acc+bias      (residual e-update)
//  MODE 1: dst  = acc+bias      (fm / wm)
//  MODE 2: g = sigmoid(acc+bias); dst = dst*(1-g) + fm*g
//  MODE 3: dst  = gelu(acc+bias)
template <int MODE>
__device__ __forceinline__ void epilogue(const v8f acc[2][4], glb_cf32 bias,
                                         lds_u16 dst, lds_cu16 fm,
                                         int n0, int lane) {
  const int lm = lane & 15;
  const int mh = (lane >> 4) << 3;                 // +8 rows for upper lanes
#pragma unroll
  for (int mt = 0; mt < 2; ++mt) {
#pragma unroll
    for (int j = 0; j < 4; ++j) {
      int   n    = n0 + 16 * j + lm;
      float bn   = bias[n];
      int   base = (mt * 16 + mh) * HDIM + n;
      lds_u16  dp = dst + base;
      lds_cu16 fp = (MODE == 2) ? (fm + base) : dst;  // fp unused unless MODE==2
#pragma unroll
      for (int r = 0; r < 8; ++r) {
        float v = acc[mt][j][r] + bn;
        if (MODE == 0) {
          dp[r * HDIM] = f2bf(bf2f(dp[r * HDIM]) + v);
        } else if (MODE == 1) {
          dp[r * HDIM] = f2bf(v);
        } else if (MODE == 2) {
          float g   = 1.0f / (1.0f + __expf(-v));
          float fmv = bf2f(fp[r * HDIM]);
          float st  = bf2f(dp[r * HDIM]);
          dp[r * HDIM] = f2bf(st * (1.0f - g) + fmv * g);
        } else {
          dp[r * HDIM] = f2bf(0.5f * v * (1.0f + erff(v * 0.70710678118654752f)));
        }
      }
    }
  }
}

__device__ __forceinline__ void zero_acc(v8f acc[2][4]) {
  const v8f z = {0.f, 0.f, 0.f, 0.f, 0.f, 0.f, 0.f, 0.f};
#pragma unroll
  for (int i = 0; i < 2; ++i)
#pragma unroll
    for (int j = 0; j < 4; ++j) acc[i][j] = z;
}

__device__ __forceinline__ void load_act(glb_cf32 g, lds_u16 dst, int tid) {
  glb_cv4f src = (glb_cv4f)g;
#pragma unroll
  for (int i = tid; i < MT * HDIM / 4; i += TPB) {
    v4f v = src[i];
    int o = i * 4;
    dst[o + 0] = f2bf(v.x); dst[o + 1] = f2bf(v.y);
    dst[o + 2] = f2bf(v.z); dst[o + 3] = f2bf(v.w);
  }
}

// Head: t = gelu(x@W1 + b1) (WMMA), then skinny out = t@W2 + b2 (lane dot+shuffle).
__device__ __forceinline__ void run_head(lds_cu16 x, lds_u16 t,
                                         glb_cu16 Wt1, glb_cf32 b1,
                                         glb_cf32 W2, glb_cf32 b2,
                                         int outc, int obase, glb_f32 out,
                                         int row0, int wv, int lane, int n0) {
  v8f acc[2][4];
  zero_acc(acc);
  gemm_panel(acc, x, Wt1, HDIM, 0, n0, lane);
  epilogue<3>(acc, b1, t, t, n0, lane);
  __syncthreads();
  for (int task = wv; task < MT * outc; task += TPB / 32) {
    int   row = task / outc, col = task - row * outc;
    float s = 0.f;
    for (int k = lane; k < HDIM; k += 32)
      s += bf2f(t[row * HDIM + k]) * W2[k * outc + col];
    for (int off = 16; off; off >>= 1) s += __shfl_down(s, off);
    if (lane == 0) out[(size_t)(row0 + row) * 7 + obase + col] = s + b2[col];
  }
  __syncthreads();
}

__global__ void __launch_bounds__(TPB, 1)
fused_kernel(const float* gsh_, const float* ge_, const float* gf_, const float* gw_,
             const uint16_t* Wt_s2e_, const uint16_t* Wt_tf_, const uint16_t* Wt_fw_,
             const uint16_t* Wt_fg_,  const uint16_t* Wt_wg_,
             const float* b_s2e_, const float* b_tf_, const float* b_fw_,
             const float* b_fg_,  const float* b_wg_,
             const uint16_t* Wt_hs1_, const float* hs_b1_,
             const float* hs_W2_,     const float* hs_b2_,
             const uint16_t* Wt_he1_, const float* he_b1_,
             const float* he_W2_,     const float* he_b2_,
             const uint16_t* Wt_hf1_, const float* hf_b1_,
             const float* hf_W2_,     const float* hf_b2_,
             const uint16_t* Wt_hw1_, const float* hw_b1_,
             const float* hw_W2_,     const float* hw_b2_,
             float* out_) {
  // Static LDS: 5 x [MT][HDIM] bf16 = 163840 B (gfx1250 addressable max).
  __shared__ uint16_t smem[5 * MT * HDIM];
  lds_u16 sSh = (lds_u16)smem;
  lds_u16 sE  = sSh + 1 * MT * HDIM;
  lds_u16 sF  = sSh + 2 * MT * HDIM;
  lds_u16 sW  = sSh + 3 * MT * HDIM;
  lds_u16 sT  = sSh + 4 * MT * HDIM;   // fm / wm / head temp

  const int tid  = threadIdx.x;
  const int row0 = blockIdx.x * MT;
  const int wv   = tid >> 5;
  const int lane = tid & 31;
  const int n0   = wv * 64;            // this wave's N slice

  // Global-AS views of all pointer args (global_load, saddr-friendly).
  glb_cu16 Wt_s2e = (glb_cu16)Wt_s2e_, Wt_tf = (glb_cu16)Wt_tf_;
  glb_cu16 Wt_fw  = (glb_cu16)Wt_fw_,  Wt_fg = (glb_cu16)Wt_fg_;
  glb_cu16 Wt_wg  = (glb_cu16)Wt_wg_;
  glb_cf32 b_s2e = (glb_cf32)b_s2e_, b_tf = (glb_cf32)b_tf_;
  glb_cf32 b_fw  = (glb_cf32)b_fw_,  b_fg = (glb_cf32)b_fg_;
  glb_cf32 b_wg  = (glb_cf32)b_wg_;
  glb_f32  out   = (glb_f32)out_;

  load_act((glb_cf32)(gsh_ + (size_t)row0 * HDIM), sSh, tid);
  load_act((glb_cf32)(ge_  + (size_t)row0 * HDIM), sE,  tid);
  load_act((glb_cf32)(gf_  + (size_t)row0 * HDIM), sF,  tid);
  load_act((glb_cf32)(gw_  + (size_t)row0 * HDIM), sW,  tid);
  __syncthreads();

  v8f acc[2][4];
  for (int l = 0; l < 5; ++l) {
    const size_t oKN  = (size_t)l * HDIM * HDIM;
    const size_t oK2N = (size_t)l * 2 * HDIM * HDIM;
    // e = e + sh @ s2e_W + b
    zero_acc(acc);
    gemm_panel(acc, sSh, Wt_s2e + oKN, HDIM, 0, n0, lane);
    epilogue<0>(acc, b_s2e + l * HDIM, sE, sE, n0, lane);
    __syncthreads();
    // fm = [sh, e] @ tf_W + b   (two K=512 panels against [N,1024] weight)
    zero_acc(acc);
    gemm_panel(acc, sSh, Wt_tf + oK2N, 2 * HDIM, 0,    n0, lane);
    gemm_panel(acc, sE,  Wt_tf + oK2N, 2 * HDIM, HDIM, n0, lane);
    epilogue<1>(acc, b_tf + l * HDIM, sT, sT, n0, lane);
    __syncthreads();
    // g = sigmoid(fm @ fg_W + b); f = f*(1-g) + fm*g
    zero_acc(acc);
    gemm_panel(acc, sT, Wt_fg + oKN, HDIM, 0, n0, lane);
    epilogue<2>(acc, b_fg + l * HDIM, sF, sT, n0, lane);
    __syncthreads();
    // wm = [e, f] @ fw_W + b  -> sT
    zero_acc(acc);
    gemm_panel(acc, sE, Wt_fw + oK2N, 2 * HDIM, 0,    n0, lane);
    gemm_panel(acc, sF, Wt_fw + oK2N, 2 * HDIM, HDIM, n0, lane);
    epilogue<1>(acc, b_fw + l * HDIM, sT, sT, n0, lane);
    __syncthreads();
    // g = sigmoid(wm @ wg_W + b); w = w*(1-g) + wm*g
    zero_acc(acc);
    gemm_panel(acc, sT, Wt_wg + oKN, HDIM, 0, n0, lane);
    epilogue<2>(acc, b_wg + l * HDIM, sW, sT, n0, lane);
    __syncthreads();
  }

  run_head(sSh, sT, (glb_cu16)Wt_hs1_, (glb_cf32)hs_b1_, (glb_cf32)hs_W2_,
           (glb_cf32)hs_b2_, 3, 0, out, row0, wv, lane, n0);
  run_head(sE,  sT, (glb_cu16)Wt_he1_, (glb_cf32)he_b1_, (glb_cf32)he_W2_,
           (glb_cf32)he_b2_, 1, 3, out, row0, wv, lane, n0);
  run_head(sF,  sT, (glb_cu16)Wt_hf1_, (glb_cf32)hf_b1_, (glb_cf32)hf_W2_,
           (glb_cf32)hf_b2_, 1, 4, out, row0, wv, lane, n0);
  run_head(sW,  sT, (glb_cu16)Wt_hw1_, (glb_cf32)hw_b1_, (glb_cf32)hw_W2_,
           (glb_cf32)hw_b2_, 2, 5, out, row0, wv, lane, n0);
}

// Weight pre-pass: fp32 [cnt][K][N] -> bf16 transposed [cnt][N][K] in d_ws.
__global__ void cvtT_kernel(const float* W_, uint16_t* o_, int K, int N, long total) {
  long idx = (long)blockIdx.x * blockDim.x + threadIdx.x;
  if (idx >= total) return;
  glb_cf32 W = (glb_cf32)W_;
  AS1 uint16_t* o = (AS1 uint16_t*)o_;
  long per = (long)K * N;
  int  m = (int)(idx / per);
  long r = idx - (long)m * per;
  int  n = (int)(r / K);
  int  k = (int)(r - (long)n * K);
  o[(long)m * per + (long)n * K + k] = f2bf(W[(long)m * per + (long)k * N + n]);
}

extern "C" void kernel_launch(void* const* d_in, const int* in_sizes, int n_in,
                              void* d_out, int out_size, void* d_ws, size_t ws_size,
                              hipStream_t stream) {
  const float* gsh   = (const float*)d_in[0];
  const float* ge    = (const float*)d_in[1];
  const float* gf    = (const float*)d_in[2];
  const float* gw    = (const float*)d_in[3];
  const float* s2e_W = (const float*)d_in[4];  const float* s2e_b = (const float*)d_in[5];
  const float* tf_W  = (const float*)d_in[6];  const float* tf_b  = (const float*)d_in[7];
  const float* fw_W  = (const float*)d_in[8];  const float* fw_b  = (const float*)d_in[9];
  const float* fg_W  = (const float*)d_in[10]; const float* fg_b  = (const float*)d_in[11];
  const float* wg_W  = (const float*)d_in[12]; const float* wg_b  = (const float*)d_in[13];
  const float* hs_W1 = (const float*)d_in[14]; const float* hs_b1 = (const float*)d_in[15];
  const float* hs_W2 = (const float*)d_in[16]; const float* hs_b2 = (const float*)d_in[17];
  const float* he_W1 = (const float*)d_in[18]; const float* he_b1 = (const float*)d_in[19];
  const float* he_W2 = (const float*)d_in[20]; const float* he_b2 = (const float*)d_in[21];
  const float* hf_W1 = (const float*)d_in[22]; const float* hf_b1 = (const float*)d_in[23];
  const float* hf_W2 = (const float*)d_in[24]; const float* hf_b2 = (const float*)d_in[25];
  const float* hw_W1 = (const float*)d_in[26]; const float* hw_b1 = (const float*)d_in[27];
  const float* hw_W2 = (const float*)d_in[28]; const float* hw_b2 = (const float*)d_in[29];

  uint16_t* ws = (uint16_t*)d_ws;   // ~20.4 MB of bf16 transposed weights
  const size_t KN = (size_t)HDIM * HDIM;
  size_t o = 0;
  size_t o_s2e = o; o += 5 * KN;
  size_t o_tf  = o; o += 5 * 2 * KN;
  size_t o_fw  = o; o += 5 * 2 * KN;
  size_t o_fg  = o; o += 5 * KN;
  size_t o_wg  = o; o += 5 * KN;
  size_t o_hs  = o; o += KN;
  size_t o_he  = o; o += KN;
  size_t o_hf  = o; o += KN;
  size_t o_hw  = o; o += KN;

  auto cvt = [&](const float* W, size_t off, int K, int N, int cnt) {
    long total  = (long)cnt * K * N;
    int  blocks = (int)((total + 255) / 256);
    cvtT_kernel<<<blocks, 256, 0, stream>>>(W, ws + off, K, N, total);
  };
  cvt(s2e_W, o_s2e, HDIM,     HDIM, 5);
  cvt(tf_W,  o_tf,  2 * HDIM, HDIM, 5);
  cvt(fw_W,  o_fw,  2 * HDIM, HDIM, 5);
  cvt(fg_W,  o_fg,  HDIM,     HDIM, 5);
  cvt(wg_W,  o_wg,  HDIM,     HDIM, 5);
  cvt(hs_W1, o_hs,  HDIM,     HDIM, 1);
  cvt(he_W1, o_he,  HDIM,     HDIM, 1);
  cvt(hf_W1, o_hf,  HDIM,     HDIM, 1);
  cvt(hw_W1, o_hw,  HDIM,     HDIM, 1);

  const int nblocks = 32768 / MT;   // 1024 row tiles
  fused_kernel<<<nblocks, TPB, 0, stream>>>(
      gsh, ge, gf, gw,
      ws + o_s2e, ws + o_tf, ws + o_fw, ws + o_fg, ws + o_wg,
      s2e_b, tf_b, fw_b, fg_b, wg_b,
      ws + o_hs, hs_b1, hs_W2, hs_b2,
      ws + o_he, he_b1, he_W2, he_b2,
      ws + o_hf, hf_b1, hf_W2, hf_b2,
      ws + o_hw, hw_b1, hw_W2, hw_b2,
      (float*)d_out);

  (void)in_sizes; (void)n_in; (void)out_size; (void)ws_size;
}
